// BidirectionalTreeLSTM_29841432773233
// MI455X (gfx1250) — matrix operand order repaired and checked by reading, not compiled, not run on hardware
//
#include <hip/hip_runtime.h>
#include <hip/hip_bf16.h>
#include <math.h>

// ---------------------------------------------------------------------------
// BidirectionalTreeLSTM on MI455X (gfx1250), wave32 + WMMA bf16.
// Output-equivalent dead-code elimination: only leaf-level MLP/iou, the
// bottom-up recurrence, the root top-down cell, and the final FC survive.
// All GEMMs run as v_wmma_f32_16x16x32_bf16 with f32 accumulation; the LSTM
// cell state c stays f32.  Weights are transposed once to K-major bf16 so A
// rows load as contiguous 8-element runs and B lanes read consecutive n at each k per the ISA layouts (CDNA5 ISA).
// ---------------------------------------------------------------------------

typedef __bf16 bf16;
typedef __attribute__((ext_vector_type(16))) __bf16 v16bf;
typedef __attribute__((ext_vector_type(8)))  __bf16 v8bf;
typedef __attribute__((ext_vector_type(8)))  float  v8f;

#define BATCH 256
#define HDIM  128
#define NTREE 1023
#define LEAF0 511
#define NLEAF 512

__device__ __forceinline__ float sigm(float x) { return 1.0f / (1.0f + __expf(-x)); }

__device__ __forceinline__ v8f wmma_bf(v16bf a, v16bf b, v8f c) {
  return __builtin_amdgcn_wmma_f32_16x16x32_bf16(false, a, false, b, (short)0, c,
                                                 false, false);
}

// A fragment (16x32 bf16, ISA 7.12.2): lane<16 holds row m=lane, K=k0+[0..7],
// k0+[16..23]; lane>=16 holds row m=lane-16, K=k0+[8..15], k0+[24..31].
__device__ __forceinline__ v16bf load_a_row_bf16(const bf16* rowp, int k0, int lane) {
  int half = lane >> 4;
  const bf16* p = rowp + k0 + half * 8;
  v8bf lo = *(const v8bf*)p;
  v8bf hi = *(const v8bf*)(p + 16);
  v16bf a;
#pragma unroll
  for (int i = 0; i < 8; ++i) { a[i] = lo[i]; a[i + 8] = hi[i]; }
  return a;
}

__device__ __forceinline__ v16bf load_a_row_f32(const float* rowp, int k0, int lane) {
  int half = lane >> 4;
  const float* p = rowp + k0 + half * 8;
  v8f lo = *(const v8f*)p;
  v8f hi = *(const v8f*)(p + 16);
  v16bf a;
#pragma unroll
  for (int i = 0; i < 8; ++i) { a[i] = (bf16)lo[i]; a[i + 8] = (bf16)hi[i]; }
  return a;
}

__device__ __forceinline__ v16bf load_a_base_bf16(const bf16* base, int ld, int k0,
                                                  int lane) {
  return load_a_row_bf16(base + (size_t)(lane & 15) * ld, k0, lane);
}

// B fragment (32x16 bf16): lane l holds column N=n0+(l&15), K=k0+8h+[0..7],
// k0+16+8h+[0..7] with h=l>>4 (wt is K-major).
__device__ __forceinline__ v16bf load_b_frag(const bf16* wt, int ldb, int k0, int n0,
                                             int lane) {
  int half = lane >> 4;
  const bf16* p = wt + (size_t)(k0 + half * 8) * ldb + n0 + (lane & 15);
  v16bf b;
#pragma unroll
  for (int i = 0; i < 8; ++i) { b[i] = p[(size_t)i * ldb]; b[i + 8] = p[(size_t)(i + 16) * ldb]; }
  return b;
}

// ---------------------------------------------------------------------------
// dst[k*R + r] = (bf16) src[r*C + k]   (weight transpose to K-major bf16)
// ---------------------------------------------------------------------------
__global__ void transpose_to_bf16(const float* __restrict__ src, bf16* __restrict__ dst,
                                  int R, int C) {
  int idx = blockIdx.x * blockDim.x + threadIdx.x;
  if (idx < R * C) {
    int r = idx / C;
    int c = idx - r * C;
    dst[(size_t)c * R + r] = (bf16)src[idx];
  }
}

// ---------------------------------------------------------------------------
// Leaf / root-TD cell: x = relu(W_mlp f + b_mlp); iou = W_iou x;
// c = sig(i)*tanh(u); h = sig(o)*tanh(c).  One wave per 16-row tile.
// ---------------------------------------------------------------------------
__global__ void input_node_kernel(const float* __restrict__ feats, int nodes_per_b,
                                  int node_base, const bf16* __restrict__ wt_mlp,
                                  const float* __restrict__ b_mlp,
                                  const bf16* __restrict__ wt_iou,
                                  const float* __restrict__ b_iou,
                                  bf16* __restrict__ h_out, float* __restrict__ c_out) {
  __shared__ __align__(16) bf16 xs[4][16][136];
  int lane = threadIdx.x & 31;
  int wave = threadIdx.x >> 5;
  int tile = blockIdx.x * (blockDim.x >> 5) + wave;
  int rows = BATCH * nodes_per_b;
  int row0 = tile * 16;
  if (row0 >= rows) return;

  int m = lane & 15;
  int half = lane >> 4;
  int grow = row0 + m;
  int b = grow / nodes_per_b;
  int j = grow - b * nodes_per_b;
  const float* arow = feats + ((size_t)b * NTREE + node_base + j) * HDIM;

  v16bf afr[4];
#pragma unroll
  for (int k = 0; k < 4; ++k) afr[k] = load_a_row_f32(arow, k * 32, lane);

  // Stage 1: x tile (16x128) -> relu -> bf16 in LDS
  for (int t = 0; t < 8; ++t) {
    v8f acc = {};
#pragma unroll
    for (int k = 0; k < 4; ++k)
      acc = wmma_bf(afr[k], load_b_frag(wt_mlp, HDIM, k * 32, t * 16, lane), acc);
    float bias = b_mlp[t * 16 + m];
#pragma unroll
    for (int r = 0; r < 8; ++r) {
      float x = acc[r] + bias;
      x = x > 0.0f ? x : 0.0f;
      xs[wave][r + 8 * half][t * 16 + m] = (bf16)x;
    }
  }

  v16bf xfr[4];
#pragma unroll
  for (int k = 0; k < 4; ++k)
    xfr[k] = load_a_base_bf16(&xs[wave][0][0], 136, k * 32, lane);

  // Stage 2: iou (16x384) + LSTM cell, c_prev = 0
  for (int t = 0; t < 8; ++t) {
    v8f ai = {}, ao = {}, au = {};
#pragma unroll
    for (int k = 0; k < 4; ++k) {
      ai = wmma_bf(xfr[k], load_b_frag(wt_iou, 384, k * 32, t * 16, lane), ai);
      ao = wmma_bf(xfr[k], load_b_frag(wt_iou, 384, k * 32, 128 + t * 16, lane), ao);
      au = wmma_bf(xfr[k], load_b_frag(wt_iou, 384, k * 32, 256 + t * 16, lane), au);
    }
    int col = t * 16 + m;
    float bi = b_iou[col], bo = b_iou[128 + col], bu = b_iou[256 + col];
#pragma unroll
    for (int r = 0; r < 8; ++r) {
      int orow = row0 + r + 8 * half;
      float cv = sigm(ai[r] + bi) * tanhf(au[r] + bu);
      float hv = sigm(ao[r] + bo) * tanhf(cv);
      c_out[(size_t)orow * HDIM + col] = cv;
      h_out[(size_t)orow * HDIM + col] = (bf16)hv;
    }
  }
}

// ---------------------------------------------------------------------------
// Bottom-up level: parents j get children 2j,2j+1 of the previous level.
// fl/fr = sig(Uf h_{l,r} + b); iou = U_iou (hl+hr); cmail = fl cl + fr cr.
// ---------------------------------------------------------------------------
__global__ void bu_level_kernel(int n_par, const bf16* __restrict__ h_prev,
                                const float* __restrict__ c_prev,
                                bf16* __restrict__ h_out, float* __restrict__ c_out,
                                const bf16* __restrict__ wt_uf,
                                const float* __restrict__ uf_b,
                                const bf16* __restrict__ wt_uiou,
                                const float* __restrict__ b_iou) {
  int lane = threadIdx.x & 31;
  int wave = threadIdx.x >> 5;
  int tile = blockIdx.x * (blockDim.x >> 5) + wave;
  int rows = BATCH * n_par;
  int row0 = tile * 16;
  if (row0 >= rows) return;

  int m = lane & 15;
  int half = lane >> 4;
  int grow = row0 + m;
  int b = grow / n_par;
  int j = grow - b * n_par;
  size_t cl = (size_t)b * (2 * n_par) + 2 * j;
  const bf16* hlrow = h_prev + cl * HDIM;
  const bf16* hrrow = h_prev + (cl + 1) * HDIM;

  v16bf al[4], ar[4], as_[4];
#pragma unroll
  for (int k = 0; k < 4; ++k) {
    al[k] = load_a_row_bf16(hlrow, k * 32, lane);
    ar[k] = load_a_row_bf16(hrrow, k * 32, lane);
#pragma unroll
    for (int i = 0; i < 16; ++i)
      as_[k][i] = (bf16)((float)al[k][i] + (float)ar[k][i]);
  }

  for (int t = 0; t < 8; ++t) {
    v8f afl = {}, afr = {}, ai = {}, ao = {}, au = {};
#pragma unroll
    for (int k = 0; k < 4; ++k) {
      v16bf bf = load_b_frag(wt_uf, HDIM, k * 32, t * 16, lane);
      afl = wmma_bf(al[k], bf, afl);
      afr = wmma_bf(ar[k], bf, afr);
      ai = wmma_bf(as_[k], load_b_frag(wt_uiou, 384, k * 32, t * 16, lane), ai);
      ao = wmma_bf(as_[k], load_b_frag(wt_uiou, 384, k * 32, 128 + t * 16, lane), ao);
      au = wmma_bf(as_[k], load_b_frag(wt_uiou, 384, k * 32, 256 + t * 16, lane), au);
    }
    int col = t * 16 + m;
    float fb = uf_b[col];
    float bi = b_iou[col], bo = b_iou[128 + col], bu = b_iou[256 + col];
#pragma unroll
    for (int r = 0; r < 8; ++r) {
      int orow = row0 + r + 8 * half;
      int bb = orow / n_par;
      int jj = orow - bb * n_par;
      size_t cli = ((size_t)bb * (2 * n_par) + 2 * jj) * HDIM + col;
      float clv = c_prev[cli];
      float crv = c_prev[cli + HDIM];
      float fl = sigm(afl[r] + fb);
      float fr = sigm(afr[r] + fb);
      float cmail = fl * clv + fr * crv;
      float cv = sigm(ai[r] + bi) * tanhf(au[r] + bu) + cmail;
      float hv = sigm(ao[r] + bo) * tanhf(cv);
      c_out[(size_t)orow * HDIM + col] = cv;
      h_out[(size_t)orow * HDIM + col] = (bf16)hv;
    }
  }
}

// ---------------------------------------------------------------------------
// Final FC: out = [c_bu_root | c_td_root] @ W_fc^T + b_fc  (256x128, K=256)
// ---------------------------------------------------------------------------
__global__ void fc_kernel(const float* __restrict__ c_bu, const float* __restrict__ c_td,
                          const bf16* __restrict__ wt_fc, const float* __restrict__ b_fc,
                          float* __restrict__ out) {
  int lane = threadIdx.x & 31;
  int wave = threadIdx.x >> 5;
  int tile = blockIdx.x * (blockDim.x >> 5) + wave;
  if (tile >= 16 * 8) return;
  int rt = tile >> 3;
  int nt = tile & 7;
  int row0 = rt * 16;
  int n0 = nt * 16;
  int m = lane & 15;
  int half = lane >> 4;
  const float* r1 = c_bu + (size_t)(row0 + m) * HDIM;
  const float* r2 = c_td + (size_t)(row0 + m) * HDIM;

  v8f acc = {};
#pragma unroll
  for (int k = 0; k < 8; ++k) {
    const float* rp = (k < 4) ? r1 : r2;
    v16bf a = load_a_row_f32(rp, (k & 3) * 32, lane);
    acc = wmma_bf(a, load_b_frag(wt_fc, HDIM, k * 32, n0, lane), acc);
  }
  int col = n0 + m;
  float bias = b_fc[col];
#pragma unroll
  for (int r = 0; r < 8; ++r)
    out[(size_t)(row0 + r + 8 * half) * HDIM + col] = acc[r] + bias;
}

// ---------------------------------------------------------------------------
extern "C" void kernel_launch(void* const* d_in, const int* in_sizes, int n_in,
                              void* d_out, int out_size, void* d_ws, size_t ws_size,
                              hipStream_t stream) {
  (void)in_sizes; (void)n_in; (void)out_size; (void)ws_size;
  const float* feats    = (const float*)d_in[0];
  const float* W_mlp    = (const float*)d_in[1];
  const float* b_mlp    = (const float*)d_in[2];
  const float* W_iou_bu = (const float*)d_in[3];
  const float* U_iou_bu = (const float*)d_in[4];
  const float* b_iou_bu = (const float*)d_in[5];
  const float* Uf_bu_w  = (const float*)d_in[6];
  const float* Uf_bu_b  = (const float*)d_in[7];
  const float* W_iou_td = (const float*)d_in[8];
  const float* b_iou_td = (const float*)d_in[10];
  const float* W_fc     = (const float*)d_in[13];
  const float* b_fc     = (const float*)d_in[14];

  char* ws = (char*)d_ws;
  size_t off = 0;
  auto take = [&](size_t bytes) {
    size_t o = off;
    off = (off + bytes + 255) & ~(size_t)255;
    return o;
  };
  const size_t HN = (size_t)BATCH * NLEAF * HDIM;  // max level state elements
  bf16*  h0       = (bf16*)(ws + take(HN * 2));
  bf16*  h1       = (bf16*)(ws + take(HN * 2));
  float* c0       = (float*)(ws + take(HN * 4));
  float* c1       = (float*)(ws + take(HN * 4));
  float* c_td     = (float*)(ws + take((size_t)BATCH * HDIM * 4));
  bf16*  wt_mlp   = (bf16*)(ws + take(128 * 128 * 2));
  bf16*  wt_ioubu = (bf16*)(ws + take(384 * 128 * 2));
  bf16*  wt_uiou  = (bf16*)(ws + take(384 * 128 * 2));
  bf16*  wt_uf    = (bf16*)(ws + take(128 * 128 * 2));
  bf16*  wt_ioutd = (bf16*)(ws + take(384 * 128 * 2));
  bf16*  wt_fc    = (bf16*)(ws + take(256 * 128 * 2));

  // Weight transposes (tiny): K-major bf16 copies.
  transpose_to_bf16<<<(128 * 128 + 255) / 256, 256, 0, stream>>>(W_mlp, wt_mlp, 128, 128);
  transpose_to_bf16<<<(384 * 128 + 255) / 256, 256, 0, stream>>>(W_iou_bu, wt_ioubu, 384, 128);
  transpose_to_bf16<<<(384 * 128 + 255) / 256, 256, 0, stream>>>(U_iou_bu, wt_uiou, 384, 128);
  transpose_to_bf16<<<(128 * 128 + 255) / 256, 256, 0, stream>>>(Uf_bu_w, wt_uf, 128, 128);
  transpose_to_bf16<<<(384 * 128 + 255) / 256, 256, 0, stream>>>(W_iou_td, wt_ioutd, 384, 128);
  transpose_to_bf16<<<(128 * 256 + 255) / 256, 256, 0, stream>>>(W_fc, wt_fc, 128, 256);

  // Leaf level: 256*512 rows -> 8192 wave-tiles, 4 waves/block.
  input_node_kernel<<<2048, 128, 0, stream>>>(feats, NLEAF, LEAF0, wt_mlp, b_mlp,
                                              wt_ioubu, b_iou_bu, h0, c0);

  // Bottom-up recurrence, levels 8..0 (ping-pong state).
  bf16* hp = h0; bf16* hc = h1;
  float* cp = c0; float* cc = c1;
  for (int lvl = 8; lvl >= 0; --lvl) {
    int n = 1 << lvl;
    int tiles = 16 * n;  // BATCH*n/16
    int blocks = (tiles + 3) / 4;
    bu_level_kernel<<<blocks, 128, 0, stream>>>(n, hp, cp, hc, cc, wt_uf, Uf_bu_b,
                                                wt_uiou, b_iou_bu);
    bf16* th = hp; hp = hc; hc = th;
    float* tc = cp; cp = cc; cc = tc;
  }
  // cp now holds c_bu root (256 x 128).

  // Root top-down cell (TD levels >=1 are dead w.r.t. the output).
  input_node_kernel<<<4, 128, 0, stream>>>(feats, 1, 0, wt_mlp, b_mlp, wt_ioutd,
                                           b_iou_td, hc /*scratch*/, c_td);

  // Final FC.
  fc_kernel<<<32, 128, 0, stream>>>(cp, c_td, wt_fc, b_fc, (float*)d_out);
}
